// Decoder_70377334112524
// MI455X (gfx1250) — matrix operand order repaired and checked by reading, compiled, not run on hardware
//
#include <hip/hip_runtime.h>
#include <math.h>

// ---------------------------------------------------------------------------
// Types for CDNA5 WMMA (wave32)
// ---------------------------------------------------------------------------
typedef __bf16 bf16;
typedef __attribute__((ext_vector_type(16))) __bf16 v16bf;
typedef __attribute__((ext_vector_type(8)))  __bf16 v8bf;
typedef __attribute__((ext_vector_type(8)))  float  v8f;

// Problem dims
constexpr int NB     = 32;    // batch
constexpr int NTENC  = 768;   // encoder steps
constexpr int NTDEC  = 400;   // decoder steps
constexpr int NMELS  = 80;
constexpr int PRE    = 256;   // prenet width
constexpr int ENC    = 512;
constexpr int RNN    = 1024;
constexpr int ATT    = 128;
constexpr int LOCF   = 32;
constexpr int LOCK   = 31;
constexpr int KPAD1  = 96;    // 80 padded up to multiple of 32

// ---------------------------------------------------------------------------
// WMMA bf16 GEMM:  C[M,N] = sum_terms A_i[M,K_i] * W_i[N,K_i]^T  (+bias,+relu)
// K is a compile-time template constant -> the K loop fully unrolls with no
// branches; the scheduler hoists load clauses across WMMAs and emits partial
// s_wait_loadcnt. NT = N-tiles per wave (2 where N%32==0): one A fragment
// feeds two independent back-to-back v_wmma ops (two B frags / two accs),
// halving A traffic and keeping the XDL pipe busy during fragment fetch.
// A fragment (16x32 bf16), per ISA 7.12.2:
//   lanes 0-15 : row m0+lane,  halves 0..7 = K k0..k0+7,  8..15 = k0+16..k0+23
//   lanes 16-31: row m0+l-16,  halves 0..7 = K k0+8..15,  8..15 = k0+24..k0+31
// B fragment (32x16): lanes 0-15 col n0+lane K k0..7,k0+16..23; lanes 16-31 K k0+8..15,k0+24..31
// ---------------------------------------------------------------------------
__device__ __forceinline__ v16bf frag16(v8bf lo, v8bf hi) {
  return __builtin_shufflevector(lo, hi, 0,1,2,3,4,5,6,7,8,9,10,11,12,13,14,15);
}

template <int K, int NT>
__device__ __forceinline__ void wmma_term(v8f (&acc)[NT],
                                          const bf16* __restrict__ A, int lda,
                                          const bf16* __restrict__ W, int ldw,
                                          int m0, int n0, int lane)
{
  const int  r  = lane & 15;
  const bool hi = lane >= 16;
  const bf16* arow  = A + (long)(m0 + r) * lda + (hi ? 8 : 0);
  const bf16* wrow0 = W + (long)(n0 + r) * ldw + (hi ? 8 : 0);
  const bf16* wrow1 = W + (long)(n0 + 16 + r) * ldw + (hi ? 8 : 0);
#pragma unroll
  for (int k0 = 0; k0 < K; k0 += 32) {
    // prefetch ~512B ahead once per 256B consumed (guards constant-fold)
    if ((k0 & 127) == 0 && k0 + 256 < K) {
      __builtin_prefetch(arow + k0 + 256, 0, 3);
      __builtin_prefetch(wrow0 + k0 + 256, 0, 3);
      if (NT == 2) __builtin_prefetch(wrow1 + k0 + 256, 0, 3);
    }
    v16bf a  = frag16(*(const v8bf*)(arow + k0), *(const v8bf*)(arow + k0 + 16));
    v16bf b0 = frag16(*(const v8bf*)(wrow0 + k0), *(const v8bf*)(wrow0 + k0 + 16));
    acc[0] = __builtin_amdgcn_wmma_f32_16x16x32_bf16(false, a, false, b0,
                                                     (short)0, acc[0], false, false);
    if (NT == 2) {
      v16bf b1 = frag16(*(const v8bf*)(wrow1 + k0), *(const v8bf*)(wrow1 + k0 + 16));
      acc[1] = __builtin_amdgcn_wmma_f32_16x16x32_bf16(false, a, false, b1,
                                                       (short)0, acc[1], false, false);
    }
  }
}

template <int K0, int K1, int K2, int NT>
__global__ void wmma_gemm(const bf16* A0, int lda0, const bf16* W0, int ldw0,
                          const bf16* A1, int lda1, const bf16* W1, int ldw1,
                          const bf16* A2, int lda2, const bf16* W2, int ldw2,
                          const float* __restrict__ bias, int act,
                          float* __restrict__ C, long ldcR, long ldcC,
                          bf16* __restrict__ Cbf, int ldcbf)
{
  const int lane = threadIdx.x;              // 32 threads = one wave32
  const int n0 = blockIdx.x * (16 * NT);
  const int m0 = blockIdx.y * 16;
  v8f acc[NT] = {};
  if (K0 > 0) wmma_term<(K0 > 0 ? K0 : 32), NT>(acc, A0, lda0, W0, ldw0, m0, n0, lane);
  if (K1 > 0) wmma_term<(K1 > 0 ? K1 : 32), NT>(acc, A1, lda1, W1, ldw1, m0, n0, lane);
  if (K2 > 0) wmma_term<(K2 > 0 ? K2 : 32), NT>(acc, A2, lda2, W2, ldw2, m0, n0, lane);

  // C/D layout (ISA 7.12.2): VGPR v -> row (m0 + v + 8*(lane>=16)), col n0+(lane&15)
  const int rbase = m0 + ((lane >= 16) ? 8 : 0);
#pragma unroll
  for (int tile = 0; tile < NT; ++tile) {
    const int col = n0 + tile * 16 + (lane & 15);
    const float bv = bias ? bias[col] : 0.f;
#pragma unroll
    for (int v = 0; v < 8; ++v) {
      float x = acc[tile][v] + bv;
      if (act) x = fmaxf(x, 0.f);
      const int row = rbase + v;
      if (C)   C[(long)row * ldcR + (long)col * ldcC] = x;
      if (Cbf) Cbf[(long)row * ldcbf + col] = (bf16)x;
    }
  }
}

// ---------------------------------------------------------------------------
// Small helper kernels
// ---------------------------------------------------------------------------
__global__ void k_f32_to_bf16(const float* __restrict__ s, bf16* __restrict__ d, long n) {
  long i = (long)blockIdx.x * blockDim.x + threadIdx.x;
  if (i < n) d[i] = (bf16)s[i];
}
__global__ void k_zero_f32(float* p, long n) {
  long i = (long)blockIdx.x * blockDim.x + threadIdx.x;
  if (i < n) p[i] = 0.f;
}
__global__ void k_zero_bf16(bf16* p, long n) {
  long i = (long)blockIdx.x * blockDim.x + threadIdx.x;
  if (i < n) p[i] = (bf16)0.f;
}
__global__ void k_pad_w(const float* __restrict__ s, bf16* __restrict__ d,
                        int rows, int ks, int kd) {
  long i = (long)blockIdx.x * blockDim.x + threadIdx.x;
  if (i < (long)rows * kd) {
    int r = (int)(i / kd), k = (int)(i % kd);
    d[i] = (k < ks) ? (bf16)s[(long)r * ks + k] : (bf16)0.f;
  }
}
// x = concat(zeros(1,B,80), transpose(decoder_inputs,(2,0,1)))[:T_DEC], padded K->96
__global__ void k_build_x(const float* __restrict__ dec, bf16* __restrict__ d) {
  long i = (long)blockIdx.x * blockDim.x + threadIdx.x;
  if (i >= (long)NTDEC * NB * KPAD1) return;
  int k = (int)(i % KPAD1);
  int b = (int)((i / KPAD1) % NB);
  int t = (int)(i / ((long)KPAD1 * NB));
  float v = 0.f;
  if (k < NMELS && t > 0) v = dec[((long)b * NMELS + k) * NTDEC + (t - 1)];
  d[i] = (bf16)v;
}
__global__ void k_addb(const float* a, const float* b, float* o, int n) {
  int i = blockIdx.x * blockDim.x + threadIdx.x;
  if (i < n) o[i] = a[i] + b[i];
}

// LSTM elementwise: gates (B,4*RNN) pre-activations -> h,c (and bf16 h)
__global__ void k_lstm(const float* __restrict__ gates, float* __restrict__ h,
                       float* __restrict__ c, bf16* __restrict__ hbf) {
  int idx = blockIdx.x * blockDim.x + threadIdx.x;   // B*RNN threads
  int b = idx >> 10, j = idx & 1023;
  const float* g = gates + (long)b * 4 * RNN;
  float gi = 1.f / (1.f + __expf(-g[j]));
  float gf = 1.f / (1.f + __expf(-g[RNN + j]));
  float gg = tanhf(g[2 * RNN + j]);
  float go = 1.f / (1.f + __expf(-g[3 * RNN + j]));
  float cn = gf * c[idx] + gi * gg;
  c[idx] = cn;
  float hn = go * tanhf(cn);
  h[idx] = hn;
  hbf[idx] = (bf16)hn;
}

// location conv + dense + tanh energy (masked). grid=(3,B), block=256
__global__ void k_energy(const float* __restrict__ aw, const float* __restrict__ awc,
                         const float* __restrict__ q, const float* __restrict__ pm,
                         const float* __restrict__ locw, const float* __restrict__ dense,
                         const float* __restrict__ vvec, const int* __restrict__ mlen,
                         float* __restrict__ energ)
{
  __shared__ float s_d[ATT * LOCF];
  __shared__ float s_w[LOCF * 2 * LOCK];
  __shared__ float s_v[ATT];
  const int tid = threadIdx.x;
  for (int i = tid; i < ATT * LOCF; i += 256) s_d[i] = dense[i];
  for (int i = tid; i < LOCF * 2 * LOCK; i += 256) s_w[i] = locw[i];
  if (tid < ATT) s_v[tid] = vvec[tid];
  __syncthreads();

  const int b = blockIdx.y;
  const int e = blockIdx.x * 256 + tid;
  if (e >= NTENC) return;
  if (e >= mlen[b]) { energ[(long)b * NTENC + e] = -1e30f; return; }

  float conv[LOCF];
  const float* awb  = aw  + (long)b * NTENC;
  const float* awcb = awc + (long)b * NTENC;
#pragma unroll 4
  for (int f = 0; f < LOCF; ++f) {
    float s = 0.f;
    for (int k = 0; k < LOCK; ++k) {
      int p = e + k - (LOCK - 1) / 2;
      if (p >= 0 && p < NTENC)
        s += s_w[f * 62 + k] * awb[p] + s_w[f * 62 + LOCK + k] * awcb[p];
    }
    conv[f] = s;
  }
  const float* qb  = q  + (long)b * ATT;
  const float* pmb = pm + ((long)b * NTENC + e) * ATT;
  float acc = 0.f;
  for (int a = 0; a < ATT; ++a) {
    float s = qb[a] + pmb[a];
#pragma unroll
    for (int f = 0; f < LOCF; ++f) s += s_d[a * LOCF + f] * conv[f];
    acc += s_v[a] * tanhf(s);
  }
  energ[(long)b * NTENC + e] = acc;
}

// softmax over T_ENC; writes aw, awc += aw, and alignment output. grid=B, block=256
__global__ void k_softmax(const float* __restrict__ energ, float* __restrict__ aw,
                          float* __restrict__ awc, float* __restrict__ out_align, int t)
{
  __shared__ float red[256];
  const int b = blockIdx.x, tid = threadIdx.x;
  const float* eb = energ + (long)b * NTENC;
  float m = -1e30f;
  for (int e = tid; e < NTENC; e += 256) m = fmaxf(m, eb[e]);
  red[tid] = m; __syncthreads();
  for (int s = 128; s > 0; s >>= 1) { if (tid < s) red[tid] = fmaxf(red[tid], red[tid + s]); __syncthreads(); }
  m = red[0]; __syncthreads();
  float sum = 0.f;
  for (int e = tid; e < NTENC; e += 256) sum += __expf(eb[e] - m);
  red[tid] = sum; __syncthreads();
  for (int s = 128; s > 0; s >>= 1) { if (tid < s) red[tid] += red[tid + s]; __syncthreads(); }
  const float inv = 1.f / red[0];
  float* awb  = aw  + (long)b * NTENC;
  float* awcb = awc + (long)b * NTENC;
  float* oa   = out_align + ((long)b * NTDEC + t) * NTENC;
  for (int e = tid; e < NTENC; e += 256) {
    float w = __expf(eb[e] - m) * inv;
    awb[e] = w; awcb[e] += w; oa[e] = w;
  }
}

// ctx[b,d] = sum_e aw[b,e]*memory[b,e,d]. grid=B, block=512
__global__ void k_ctx(const float* __restrict__ aw, const float* __restrict__ mem,
                      float* __restrict__ ctx, bf16* __restrict__ ctxbf)
{
  __shared__ float s_aw[NTENC];
  const int b = blockIdx.x, tid = threadIdx.x;
  for (int e = tid; e < NTENC; e += 512) s_aw[e] = aw[(long)b * NTENC + e];
  __syncthreads();
  const float* mb = mem + (long)b * NTENC * ENC;
  float acc = 0.f;
  for (int e = 0; e < NTENC; ++e) acc += s_aw[e] * mb[(long)e * ENC + tid];
  ctx[(long)b * ENC + tid] = acc;
  ctxbf[(long)b * ENC + tid] = (bf16)acc;
}

// gate = sigmoid([dh|ctx] . gate_w + gate_b). block=32 (one thread per batch row)
__global__ void k_gate(const float* __restrict__ dh, const float* __restrict__ ctx,
                       const float* __restrict__ gw, const float* __restrict__ gb,
                       float* __restrict__ out_gate, int t)
{
  const int b = threadIdx.x;
  float s = gb[0];
  const float* dhb = dh  + (long)b * RNN;
  const float* cb  = ctx + (long)b * ENC;
  for (int j = 0; j < RNN; ++j) s += gw[j] * dhb[j];
  for (int j = 0; j < ENC; ++j) s += gw[RNN + j] * cb[j];
  out_gate[(long)b * NTDEC + t] = 1.f / (1.f + __expf(-s));
}

// ---------------------------------------------------------------------------
// Host orchestration
// ---------------------------------------------------------------------------
extern "C" void kernel_launch(void* const* d_in, const int* in_sizes, int n_in,
                              void* d_out, int out_size, void* d_ws, size_t ws_size,
                              hipStream_t stream)
{
  (void)in_sizes; (void)n_in; (void)out_size; (void)ws_size;

  const float* memory      = (const float*)d_in[0];
  const float* dec_in      = (const float*)d_in[1];
  const int*   mlen        = (const int*)  d_in[2];
  const float* pre_w1      = (const float*)d_in[3];
  const float* pre_b1      = (const float*)d_in[4];
  const float* pre_w2      = (const float*)d_in[5];
  const float* pre_b2      = (const float*)d_in[6];
  const float* arnn_w_ih   = (const float*)d_in[7];
  const float* arnn_w_hh   = (const float*)d_in[8];
  const float* arnn_b_ih   = (const float*)d_in[9];
  const float* arnn_b_hh   = (const float*)d_in[10];
  const float* attn_q_w    = (const float*)d_in[11];
  const float* attn_mem_w  = (const float*)d_in[12];
  const float* attn_v      = (const float*)d_in[13];
  const float* loc_conv_w  = (const float*)d_in[14];
  const float* loc_dense_w = (const float*)d_in[15];
  const float* drnn_w_ih   = (const float*)d_in[16];
  const float* drnn_w_hh   = (const float*)d_in[17];
  const float* drnn_b_ih   = (const float*)d_in[18];
  const float* drnn_b_hh   = (const float*)d_in[19];
  const float* proj_w      = (const float*)d_in[20];
  const float* proj_b      = (const float*)d_in[21];
  const float* gate_w      = (const float*)d_in[22];
  const float* gate_b      = (const float*)d_in[23];

  float* out_mel   = (float*)d_out;                                // (B,80,400)
  float* out_gate  = out_mel  + (size_t)NB * NMELS * NTDEC;        // (B,400)
  float* out_align = out_gate + (size_t)NB * NTDEC;                // (B,400,768)

  // Workspace carve (256B aligned)
  char* cur = (char*)d_ws;
  auto alloc = [&](size_t bytes) -> void* {
    void* r = cur; cur += (bytes + 255) & ~(size_t)255; return r;
  };
  bf16* wpre1 = (bf16*)alloc(2ull * PRE * KPAD1);
  bf16* wpre2 = (bf16*)alloc(2ull * PRE * PRE);
  bf16* waih  = (bf16*)alloc(2ull * 4 * RNN * (PRE + ENC));       // 4096 x 768
  bf16* wahh  = (bf16*)alloc(2ull * 4 * RNN * RNN);
  bf16* wq    = (bf16*)alloc(2ull * ATT * RNN);
  bf16* wmem  = (bf16*)alloc(2ull * ATT * ENC);
  bf16* wdih  = (bf16*)alloc(2ull * 4 * RNN * (RNN + ENC));       // 4096 x 1536
  bf16* wdhh  = (bf16*)alloc(2ull * 4 * RNN * RNN);
  bf16* wproj = (bf16*)alloc(2ull * NMELS * (RNN + ENC));         // 80 x 1536
  bf16* membf = (bf16*)alloc(2ull * NB * NTENC * ENC);
  bf16* xpre  = (bf16*)alloc(2ull * NTDEC * NB * KPAD1);
  bf16* pn1   = (bf16*)alloc(2ull * NTDEC * NB * PRE);
  bf16* pn2   = (bf16*)alloc(2ull * NTDEC * NB * PRE);
  bf16* ahbf  = (bf16*)alloc(2ull * NB * RNN);
  bf16* dhbf  = (bf16*)alloc(2ull * NB * RNN);
  bf16* ctxbf = (bf16*)alloc(2ull * NB * ENC);
  float* pm   = (float*)alloc(4ull * NB * NTENC * ATT);
  float* gA   = (float*)alloc(4ull * NB * 4 * RNN);
  float* gD   = (float*)alloc(4ull * NB * 4 * RNN);
  float* ah   = (float*)alloc(4ull * NB * RNN);
  float* ac   = (float*)alloc(4ull * NB * RNN);
  float* dh   = (float*)alloc(4ull * NB * RNN);
  float* dc   = (float*)alloc(4ull * NB * RNN);
  float* ctx  = (float*)alloc(4ull * NB * ENC);
  float* qbuf = (float*)alloc(4ull * NB * ATT);
  float* energ= (float*)alloc(4ull * NB * NTENC);
  float* aw   = (float*)alloc(4ull * NB * NTENC);
  float* awc  = (float*)alloc(4ull * NB * NTENC);
  float* ba   = (float*)alloc(4ull * 4 * RNN);
  float* bd   = (float*)alloc(4ull * 4 * RNN);

  auto cvt = [&](const float* s, bf16* d, long n) {
    k_f32_to_bf16<<<dim3((unsigned)((n + 255) / 256)), dim3(256), 0, stream>>>(s, d, n);
  };
  auto zf = [&](float* p, long n) {
    k_zero_f32<<<dim3((unsigned)((n + 255) / 256)), dim3(256), 0, stream>>>(p, n);
  };
  auto zb = [&](bf16* p, long n) {
    k_zero_bf16<<<dim3((unsigned)((n + 255) / 256)), dim3(256), 0, stream>>>(p, n);
  };

  // ---- one-time setup (re-done each call: deterministic & idempotent) ----
  k_pad_w<<<dim3((PRE * KPAD1 + 255) / 256), dim3(256), 0, stream>>>(pre_w1, wpre1, PRE, NMELS, KPAD1);
  cvt(pre_w2, wpre2, (long)PRE * PRE);
  cvt(arnn_w_ih, waih, (long)4 * RNN * (PRE + ENC));
  cvt(arnn_w_hh, wahh, (long)4 * RNN * RNN);
  cvt(attn_q_w, wq, (long)ATT * RNN);
  cvt(attn_mem_w, wmem, (long)ATT * ENC);
  cvt(drnn_w_ih, wdih, (long)4 * RNN * (RNN + ENC));
  cvt(drnn_w_hh, wdhh, (long)4 * RNN * RNN);
  cvt(proj_w, wproj, (long)NMELS * (RNN + ENC));
  cvt(memory, membf, (long)NB * NTENC * ENC);
  k_build_x<<<dim3((unsigned)(((long)NTDEC * NB * KPAD1 + 255) / 256)), dim3(256), 0, stream>>>(dec_in, xpre);
  k_addb<<<dim3((4 * RNN + 255) / 256), dim3(256), 0, stream>>>(arnn_b_ih, arnn_b_hh, ba, 4 * RNN);
  k_addb<<<dim3((4 * RNN + 255) / 256), dim3(256), 0, stream>>>(drnn_b_ih, drnn_b_hh, bd, 4 * RNN);

  zf(ah, NB * RNN); zf(ac, NB * RNN); zf(dh, NB * RNN); zf(dc, NB * RNN);
  zf(ctx, NB * ENC); zf(aw, NB * NTENC); zf(awc, NB * NTENC);
  zb(ahbf, NB * RNN); zb(dhbf, NB * RNN); zb(ctxbf, NB * ENC);

  const bf16* nA = nullptr; const bf16* nW = nullptr;
  const dim3 W32(32);

  // prenet layer 1: (12800 x 96) @ (96 -> 256), relu, bf16 out
  wmma_gemm<KPAD1, 0, 0, 2><<<dim3(PRE / 32, (NTDEC * NB) / 16), W32, 0, stream>>>(
      xpre, KPAD1, wpre1, KPAD1,  nA, 0, nW, 0,  nA, 0, nW, 0,
      pre_b1, 1, (float*)nullptr, 0, 0, pn1, PRE);
  // prenet layer 2: (12800 x 256) @ (256 -> 256), relu, bf16 out
  wmma_gemm<PRE, 0, 0, 2><<<dim3(PRE / 32, (NTDEC * NB) / 16), W32, 0, stream>>>(
      pn1, PRE, wpre2, PRE,  nA, 0, nW, 0,  nA, 0, nW, 0,
      pre_b2, 1, (float*)nullptr, 0, 0, pn2, PRE);
  // processed_memory: (24576 x 512) @ (512 -> 128), f32 out
  wmma_gemm<ENC, 0, 0, 2><<<dim3(ATT / 32, (NB * NTENC) / 16), W32, 0, stream>>>(
      membf, ENC, wmem, ENC,  nA, 0, nW, 0,  nA, 0, nW, 0,
      (const float*)nullptr, 0, pm, ATT, 1, (bf16*)nullptr, 0);

  // ---- sequential decoder scan ----
  for (int t = 0; t < NTDEC; ++t) {
    // attention LSTM gates: [prenet|ctx] @ Wih^T + ah @ Whh^T + (b_ih+b_hh)
    wmma_gemm<PRE, ENC, RNN, 2><<<dim3(4 * RNN / 32, NB / 16), W32, 0, stream>>>(
        pn2 + (long)t * NB * PRE, PRE, waih, PRE + ENC,
        ctxbf, ENC, waih + PRE, PRE + ENC,
        ahbf, RNN, wahh, RNN,
        ba, 0, gA, 4 * RNN, 1, (bf16*)nullptr, 0);
    k_lstm<<<dim3(NB * RNN / 256), dim3(256), 0, stream>>>(gA, ah, ac, ahbf);

    // attention query q = ah @ attn_q_w^T
    wmma_gemm<RNN, 0, 0, 2><<<dim3(ATT / 32, NB / 16), W32, 0, stream>>>(
        ahbf, RNN, wq, RNN,  nA, 0, nW, 0,  nA, 0, nW, 0,
        (const float*)nullptr, 0, qbuf, ATT, 1, (bf16*)nullptr, 0);

    // energies (uses previous aw/awc), then softmax (updates aw, awc, aligns)
    k_energy<<<dim3(NTENC / 256, NB), dim3(256), 0, stream>>>(
        aw, awc, qbuf, pm, loc_conv_w, loc_dense_w, attn_v, mlen, energ);
    k_softmax<<<dim3(NB), dim3(256), 0, stream>>>(energ, aw, awc, out_align, t);
    k_ctx<<<dim3(NB), dim3(512), 0, stream>>>(aw, memory, ctx, ctxbf);

    // decoder LSTM gates: [ah|ctx] @ Wih^T + dh @ Whh^T + bias
    wmma_gemm<RNN, ENC, RNN, 2><<<dim3(4 * RNN / 32, NB / 16), W32, 0, stream>>>(
        ahbf, RNN, wdih, RNN + ENC,
        ctxbf, ENC, wdih + RNN, RNN + ENC,
        dhbf, RNN, wdhh, RNN,
        bd, 0, gD, 4 * RNN, 1, (bf16*)nullptr, 0);
    k_lstm<<<dim3(NB * RNN / 256), dim3(256), 0, stream>>>(gD, dh, dc, dhbf);

    // mel projection straight into output (B,80,400): strided store (N=80 -> NT=1)
    wmma_gemm<RNN, ENC, 0, 1><<<dim3(NMELS / 16, NB / 16), W32, 0, stream>>>(
        dhbf, RNN, wproj, RNN + ENC,
        ctxbf, ENC, wproj + RNN, RNN + ENC,
        nA, 0, nW, 0,
        proj_b, 0, out_mel + t, (long)NMELS * NTDEC, (long)NTDEC,
        (bf16*)nullptr, 0);

    k_gate<<<dim3(1), dim3(NB), 0, stream>>>(dh, ctx, gate_w, gate_b, out_gate, t);
  }
}
